// GCN_60868276519239
// MI455X (gfx1250) — matrix operand order repaired and checked by reading, not compiled, not run on hardware
//
#include <hip/hip_runtime.h>

// ---------------------------------------------------------------------------
// CDNA5 (gfx1250) GCN: 3x (WMMA GEMM -> normalized scatter-add aggregation)
// wave32 everywhere; bf16 WMMA with hi/lo split for ~fp32 GEMM accuracy;
// hardware global_atomic_add_f32 (device scope) for the L2-resident scatter.
// ---------------------------------------------------------------------------

typedef __attribute__((ext_vector_type(16))) __bf16 v16bf;
typedef __attribute__((ext_vector_type(8)))  float  v8f;

__device__ __forceinline__ void gatomic_fadd(float* p, float v) {
    // Non-returning f32 atomic add at device scope (L2 atomic units).
    asm volatile("global_atomic_add_f32 %0, %1, off scope:SCOPE_DEV"
                 :: "v"(p), "v"(v) : "memory");
}

// ---- degree / normalization -------------------------------------------------
__global__ void k_fill1(float* deg, int N) {
    int i = blockIdx.x * 256 + threadIdx.x;
    if (i < N) deg[i] = 1.0f;                       // self-loop contribution
}
__global__ void k_deg(const int* __restrict__ col, int E, float* deg) {
    int i = blockIdx.x * 256 + threadIdx.x;
    if (i < E) gatomic_fadd(&deg[col[i]], 1.0f);
}
__global__ void k_rsqrt(float* d, int N) {
    int i = blockIdx.x * 256 + threadIdx.x;
    if (i < N) d[i] = rsqrtf(d[i]);                 // deg >= 1 always
}

// ---- GEMM: Y[N x DN] = X[N x 128] * W[128 x DN], bf16 WMMA hi/lo split ------
// One wave -> one 16x16 output tile. K = 128 in 4 steps of 32.
template<int DN>
__launch_bounds__(256)
__global__ void k_gemm_wmma(const float* __restrict__ X, const float* __restrict__ Wt,
                            float* __restrict__ Y, int nRows) {
    constexpr int NT = DN / 16;
    const int wave = (blockIdx.x * 256 + threadIdx.x) >> 5;
    const int lane = threadIdx.x & 31;
    const int mt = wave / NT;
    const int nt = wave - mt * NT;
    if (mt * 16 >= nRows) return;

    const int r = lane & 15;           // row-in-tile (A), col-in-tile (B/C)
    const int h = lane >> 4;           // lane half selects K sub-range
    const float* arow = X + (size_t)(mt * 16 + r) * 128;
    const int cn = nt * 16 + r;

    v8f c = {};
    #pragma unroll
    for (int kb = 0; kb < 128; kb += 32) {
        // A 16x32 bf16 layout: lane half h, VGPR v<4: K=8h+2v+e ; v>=4: K=16+8h+...
        // => element j in [0,8)  maps to K = kb + 8h + j        (contiguous)
        //    element j in [8,16) maps to K = kb + 16 + 8h + (j-8) (contiguous)
        float4 a0 = *(const float4*)(arow + kb + 8 * h);
        float4 a1 = *(const float4*)(arow + kb + 8 * h + 4);
        float4 a2 = *(const float4*)(arow + kb + 16 + 8 * h);
        float4 a3 = *(const float4*)(arow + kb + 16 + 8 * h + 4);
        float af[16] = {a0.x, a0.y, a0.z, a0.w, a1.x, a1.y, a1.z, a1.w,
                        a2.x, a2.y, a2.z, a2.w, a3.x, a3.y, a3.z, a3.w};

        v16bf ahi, alo, bhi, blo;
        #pragma unroll
        for (int j = 0; j < 16; ++j) {
            float f = af[j];
            __bf16 hi = (__bf16)f;
            ahi[j] = hi;
            alo[j] = (__bf16)(f - (float)hi);
        }
        // B 32x16 bf16 layout: element j<8 -> K = 8h + j; j>=8 -> K = 16 + 8h + (j-8), col = cn
        #pragma unroll
        for (int j = 0; j < 16; ++j) {
            float f = Wt[(size_t)(kb + (j < 8 ? 8 * h + j : 16 + 8 * h + (j - 8))) * DN + cn];
            __bf16 hi = (__bf16)f;
            bhi[j] = hi;
            blo[j] = (__bf16)(f - (float)hi);
        }
        // 3-product split: (Ahi+Alo)(Bhi+Blo) ~= AhiBhi + AloBhi + AhiBlo
        c = __builtin_amdgcn_wmma_f32_16x16x32_bf16(false, ahi, false, bhi,
                                                    (short)0, c, false, false);
        c = __builtin_amdgcn_wmma_f32_16x16x32_bf16(false, alo, false, bhi,
                                                    (short)0, c, false, false);
        c = __builtin_amdgcn_wmma_f32_16x16x32_bf16(false, ahi, false, blo,
                                                    (short)0, c, false, false);
    }
    // C/D layout: VGPR i -> row M = 8h + i, col N = r
    #pragma unroll
    for (int i = 0; i < 8; ++i)
        Y[(size_t)(mt * 16 + 8 * h + i) * DN + cn] = c[i];
}

// ---- aggregation ------------------------------------------------------------
// agg[n,f] = dinv[n]^2 * xw[n,f] + bias[f]   (self-loop term + bias folded in)
template<int D>
__global__ void k_init_agg(const float* __restrict__ xw, const float* __restrict__ dinv,
                           const float* __restrict__ bias, float* __restrict__ agg, int N) {
    long long idx = (long long)blockIdx.x * 256 + threadIdx.x;
    if (idx >= (long long)N * D) return;
    int n = (int)(idx / D);
    int f = (int)(idx - (long long)n * D);
    float di = dinv[n];
    agg[idx] = di * di * xw[idx] + bias[f];
}

// One wave per edge: gather row r (L2-resident), scale, atomic scatter to col c.
template<int D>
__launch_bounds__(256)
__global__ void k_scatter(const float* __restrict__ xw, const int* __restrict__ row,
                          const int* __restrict__ col, const float* __restrict__ dinv,
                          float* __restrict__ agg, int E) {
    static_assert(D == 128 || D == 64, "D must be 64 or 128");
    const int e = (blockIdx.x * 256 + threadIdx.x) >> 5;
    const int lane = threadIdx.x & 31;
    if (e >= E) return;
    const int r = row[e];
    const int c = col[e];
    const float nrm = dinv[r] * dinv[c];
    constexpr int V = D / 32;          // floats per lane: 4 (D=128) or 2 (D=64)
    const float* src = xw + (size_t)r * D + lane * V;
    float* dst = agg + (size_t)c * D + lane * V;
    if (V == 4) {
        float4 v = *(const float4*)src;
        gatomic_fadd(dst + 0, nrm * v.x);
        gatomic_fadd(dst + 1, nrm * v.y);
        gatomic_fadd(dst + 2, nrm * v.z);
        gatomic_fadd(dst + 3, nrm * v.w);
    } else {
        float2 v = *(const float2*)src;
        gatomic_fadd(dst + 0, nrm * v.x);
        gatomic_fadd(dst + 1, nrm * v.y);
    }
}

__global__ void k_relu(float* a, long long n) {
    long long i = (long long)blockIdx.x * 256 + threadIdx.x;
    if (i < n) a[i] = fmaxf(a[i], 0.0f);
}

// ---------------------------------------------------------------------------
static inline int cdiv(long long a, long long b) { return (int)((a + b - 1) / b); }

extern "C" void kernel_launch(void* const* d_in, const int* in_sizes, int n_in,
                              void* d_out, int out_size, void* d_ws, size_t ws_size,
                              hipStream_t stream) {
    (void)n_in; (void)out_size; (void)ws_size;
    const float* x  = (const float*)d_in[0];
    const int*   ei = (const int*)d_in[1];
    const float* W1 = (const float*)d_in[2];
    const float* b1 = (const float*)d_in[3];
    const float* W2 = (const float*)d_in[4];
    const float* b2 = (const float*)d_in[5];
    const float* W3 = (const float*)d_in[6];
    const float* b3 = (const float*)d_in[7];
    float* out = (float*)d_out;

    const int N = in_sizes[0] / 128;   // 100000
    const int E = in_sizes[1] / 2;     // 1600000
    const int* row = ei;               // src
    const int* col = ei + E;           // dst

    // Workspace: dinv (N) + two N x 128 ping-pong buffers (~103 MB total)
    float* ws   = (float*)d_ws;
    size_t Npad = ((size_t)N + 63) & ~(size_t)63;
    float* dinv = ws;
    float* bufA = ws + Npad;
    float* bufB = bufA + (size_t)N * 128;

    const int tN   = cdiv(N, 256);
    const int tE   = cdiv(E, 256);
    const int tND  = cdiv((long long)N * 128, 256);
    const int tND3 = cdiv((long long)N * 64, 256);
    const int eWB  = cdiv(E, 8);                 // 8 waves (edges) per block
    const int mt   = (N + 15) / 16;
    const int g128 = cdiv((long long)mt * 8, 8); // waves = mt * (128/16)
    const int g64  = cdiv((long long)mt * 4, 8); // waves = mt * (64/16)

    // --- normalization: dinv = rsqrt(1 + in-degree) ---
    k_fill1<<<tN, 256, 0, stream>>>(dinv, N);
    k_deg<<<tE, 256, 0, stream>>>(col, E, dinv);
    k_rsqrt<<<tN, 256, 0, stream>>>(dinv, N);

    // --- layer 1: h1 = relu(Ahat (x W1) + b1) ---
    k_gemm_wmma<128><<<g128, 256, 0, stream>>>(x, W1, bufA, N);
    k_init_agg<128><<<tND, 256, 0, stream>>>(bufA, dinv, b1, bufB, N);
    k_scatter<128><<<eWB, 256, 0, stream>>>(bufA, row, col, dinv, bufB, E);
    k_relu<<<tND, 256, 0, stream>>>(bufB, (long long)N * 128);

    // --- layer 2: h2 = relu(Ahat (h1 W2) + b2) ---
    k_gemm_wmma<128><<<g128, 256, 0, stream>>>(bufB, W2, bufA, N);
    k_init_agg<128><<<tND, 256, 0, stream>>>(bufA, dinv, b2, bufB, N);
    k_scatter<128><<<eWB, 256, 0, stream>>>(bufA, row, col, dinv, bufB, E);
    k_relu<<<tND, 256, 0, stream>>>(bufB, (long long)N * 128);

    // --- layer 3: out = Ahat (h2 W3) + b3 (scatter straight into d_out) ---
    k_gemm_wmma<64><<<g64, 256, 0, stream>>>(bufB, W3, bufA, N);
    k_init_agg<64><<<tND3, 256, 0, stream>>>(bufA, dinv, b3, out, N);
    k_scatter<64><<<eWB, 256, 0, stream>>>(bufA, row, col, dinv, out, E);
}
